// HeteroGNN_19963007992140
// MI455X (gfx1250) — matrix operand order repaired and checked by reading, not compiled, not run on hardware
//
#include <hip/hip_runtime.h>

typedef __attribute__((ext_vector_type(16))) __bf16 v16bf;
typedef __attribute__((ext_vector_type(8)))  float  v8f;

union BF16Frag { uint4 q[2]; v16bf v; };

__device__ __forceinline__ unsigned short f2bf(float f) {
  unsigned int u = __float_as_uint(f);
  u += 0x7FFFu + ((u >> 16) & 1u);           // round-to-nearest-even
  return (unsigned short)(u >> 16);
}
__device__ __forceinline__ float bf2f(unsigned short h) {
  return __uint_as_float(((unsigned int)h) << 16);
}

// ---------------------------------------------------------------------------
// Pack a f32 [256, N] weight (optionally summing up to 3 sources) into the
// contiguous-k bf16 B-fragment layout used by the GEMM: frag (kt,nt) of 512 bf16;
// lane<16: col=lane, k=kt*32+i ; lane>=16: col=lane-16, k=kt*32+16+i.
// Npad is a multiple of 64 so the GEMM never needs tile guards on loads.
// ---------------------------------------------------------------------------
__global__ __launch_bounds__(256)
void pack_w_kernel(const float* __restrict__ W0, const float* __restrict__ W1,
                   const float* __restrict__ W2, int N, int Npad,
                   unsigned short* __restrict__ out) {
  int tid = blockIdx.x * 256 + threadIdx.x;
  int ntiles = Npad >> 4;
  int total = 8 * ntiles * 512;               // K = 256 -> 8 k-tiles of 32
  if (tid >= total) return;
  int i    = tid & 15;
  int lane = (tid >> 4) & 31;
  int f    = tid >> 9;
  int kt = f / ntiles;
  int nt = f - kt * ntiles;
  int nl = lane & 15;
  int kl = (lane < 16) ? i : (16 + i);
  int col = nt * 16 + nl;
  int row = kt * 32 + kl;
  float v = 0.f;
  if (col < N) {
    v = W0[(size_t)row * N + col];
    if (W1) v += W1[(size_t)row * N + col];
    if (W2) v += W2[(size_t)row * N + col];
  }
  out[tid] = f2bf(v);
}

__global__ __launch_bounds__(256)
void comb_bias_kernel(const float* __restrict__ b0, const float* __restrict__ b1,
                      const float* __restrict__ b2, float* __restrict__ out) {
  int i = threadIdx.x;                         // 256 threads
  float v = b0[i];
  if (b1) v += b1[i];
  if (b2) v += b2[i];
  out[i] = v;
}

__global__ __launch_bounds__(256)
void f32_to_bf16_kernel(const float* __restrict__ x, unsigned short* __restrict__ y, int n) {
  int i = blockIdx.x * 256 + threadIdx.x;
  if (i < n) y[i] = f2bf(x[i]);
}

// One wave per edge: gather 256 bf16 feats of src node, scatter-add f32 into agg[dst].
__global__ __launch_bounds__(256)
void gather_kernel(const unsigned short* __restrict__ xsrc,
                   const int* __restrict__ src, const int* __restrict__ dst, int E,
                   float* __restrict__ agg, float* __restrict__ cnt) {
  int e = blockIdx.x * 8 + (threadIdx.x >> 5);
  int lane = threadIdx.x & 31;
  if (e >= E) return;
  int s = src[e], d = dst[e];
  const unsigned short* sr = xsrc + (size_t)s * 256 + lane * 8;
  uint4 q = *(const uint4*)sr;                 // 8 bf16, 16B coalesced
  float* ar = agg + (size_t)d * 256 + lane * 8;
  unsigned int w[4] = {q.x, q.y, q.z, q.w};
#pragma unroll
  for (int j = 0; j < 4; ++j) {
    atomicAdd(ar + 2 * j,     bf2f((unsigned short)(w[j] & 0xFFFFu)));
    atomicAdd(ar + 2 * j + 1, bf2f((unsigned short)(w[j] >> 16)));
  }
  if (lane == 0) atomicAdd(cnt + d, 1.0f);
}

__global__ __launch_bounds__(256)
void mean_kernel(const float* __restrict__ agg, const float* __restrict__ cnt,
                 unsigned short* __restrict__ meanBF, int nrows) {
  int i = blockIdx.x * 256 + threadIdx.x;
  if (i >= (nrows << 8)) return;
  float c = cnt[i >> 8];
  meanBF[i] = f2bf(agg[i] / fmaxf(c, 1.0f));
}

__global__ __launch_bounds__(256)
void relu_bf16_kernel(const float* __restrict__ acc, unsigned short* __restrict__ out, int n) {
  int i = blockIdx.x * 256 + threadIdx.x;
  if (i < n) out[i] = f2bf(fmaxf(acc[i], 0.f));
}

// ---------------------------------------------------------------------------
// WMMA bf16 GEMM:  Out[M,Nout] (+)= A[M,256] @ Wpacked[256,Npad] (+ bias)
// Block = 256 thr (8 waves); block tile 256(M) x 64(N); wave tile 32 x 64
// (2 M-subtiles x 4 N-subtiles = 8 accumulators).
// Fully-unrolled 32-step (kt,t) pipeline with double-buffered A and B
// fragments: the load for step s+1 is issued before the WMMAs of step s,
// so every WMMA pair has loads in flight (partial loadcnt waits).
// Npad (= NwTiles*16) is a multiple of 64; garbage pad tiles never stored.
// ---------------------------------------------------------------------------
__global__ __launch_bounds__(256)
void gemm_bf16_kernel(const unsigned short* __restrict__ A,
                      const unsigned short* __restrict__ Wp,
                      const float* __restrict__ bias,
                      float* __restrict__ Out,
                      int M, int NwTiles, int Nout, int accum) {
  const int lane    = threadIdx.x & 31;
  const int wave    = threadIdx.x >> 5;
  const int m0      = blockIdx.x * 256 + wave * 32;
  const int nBase   = blockIdx.y * 64;
  const int nt0     = nBase >> 4;
  const int halfSel = lane >> 4;               // 0: lanes 0-15, 1: lanes 16-31
  const int l15     = lane & 15;

  v8f acc[2][4];
  int ncol[4];
#pragma unroll
  for (int t = 0; t < 4; ++t) ncol[t] = nBase + t * 16 + l15;

#pragma unroll
  for (int mi = 0; mi < 2; ++mi)
#pragma unroll
    for (int t = 0; t < 4; ++t) {
      float b = 0.f;
      if (bias && ncol[t] < Nout) b = bias[ncol[t]];
      v8f c;
#pragma unroll
      for (int v = 0; v < 8; ++v) c[v] = b;
      if (accum && ncol[t] < Nout) {
#pragma unroll
        for (int v = 0; v < 8; ++v) {
          int r = m0 + mi * 16 + v + halfSel * 8;
          if (r < M) c[v] = Out[(size_t)r * Nout + ncol[t]];
        }
      }
      acc[mi][t] = c;
    }

  // A fragment bases: lanes 0-15 read K {0..15}; lanes 16-31 {16..31}
  const unsigned short* aptr0 = A + (size_t)(m0 + l15) * 256 + halfSel * 16;
  const unsigned short* aptr1 = aptr0 + 16 * 256;
  const unsigned short* bbase = Wp + (size_t)lane * 16;

  BF16Frag af[2][2];                           // [mi][buffer]
  BF16Frag bf[2];                              // [buffer]

  auto loadA = [&](int buf, int kt) {
    af[0][buf].q[0] = *(const uint4*)(aptr0 + kt * 32);
    af[0][buf].q[1] = *(const uint4*)(aptr0 + kt * 32 + 8);
    af[1][buf].q[0] = *(const uint4*)(aptr1 + kt * 32);
    af[1][buf].q[1] = *(const uint4*)(aptr1 + kt * 32 + 8);
  };
  auto loadB = [&](int buf, int kt, int t) {
    const unsigned short* p = bbase + ((size_t)(kt * NwTiles + nt0 + t) << 9);
    bf[buf].q[0] = *(const uint4*)(p);
    bf[buf].q[1] = *(const uint4*)(p + 8);
  };

  loadA(0, 0);
  loadB(0, 0, 0);
#pragma unroll
  for (int s = 0; s < 32; ++s) {               // s = kt*4 + t, fully unrolled
    const int kt = s >> 2;
    const int t  = s & 3;
    if (s + 1 < 32) loadB((s + 1) & 1, (s + 1) >> 2, (s + 1) & 3);
    if (t == 0 && kt + 1 < 8) loadA((kt + 1) & 1, kt + 1);  // early A prefetch
    acc[0][t] = __builtin_amdgcn_wmma_f32_16x16x32_bf16(
        false, af[0][kt & 1].v, false, bf[s & 1].v, (short)0, acc[0][t], false, false);
    acc[1][t] = __builtin_amdgcn_wmma_f32_16x16x32_bf16(
        false, af[1][kt & 1].v, false, bf[s & 1].v, (short)0, acc[1][t], false, false);
  }

#pragma unroll
  for (int mi = 0; mi < 2; ++mi)
#pragma unroll
    for (int t = 0; t < 4; ++t) {
      if (ncol[t] >= Nout) continue;
#pragma unroll
      for (int v = 0; v < 8; ++v) {
        int r = m0 + mi * 16 + v + halfSel * 8;
        if (r < M) Out[(size_t)r * Nout + ncol[t]] = acc[mi][t][v];
      }
    }
}

// ---------------------------------------------------------------------------
extern "C" void kernel_launch(void* const* d_in, const int* in_sizes, int n_in,
                              void* d_out, int out_size, void* d_ws, size_t ws_size,
                              hipStream_t stream) {
  const float* xin[4] = {(const float*)d_in[0], (const float*)d_in[1],
                         (const float*)d_in[2], (const float*)d_in[3]};
  int nNodes[4], pads[4];
  for (int t = 0; t < 4; ++t) {
    nNodes[t] = in_sizes[t] / 256;
    pads[t]   = ((nNodes[t] + 255) / 256) * 256;   // M padded to block tile
  }
  const float* Wl   = (const float*)d_in[18];
  const float* bl   = (const float*)d_in[19];
  const float* Wr   = (const float*)d_in[20];
  const float* Wout = (const float*)d_in[21];
  const float* bout = (const float*)d_in[22];

  // relation r: src-type, dst-type  (edge arrays at d_in[4+2r], d_in[5+2r])
  const int relSrcT[7] = {0, 1, 0, 1, 2, 0, 3};
  const int relDstT[7] = {0, 0, 1, 2, 1, 3, 0};
  // relations feeding each destination type (for Wr / bias combining)
  const int relOfT[4][3] = {{0, 1, 6}, {2, 4, -1}, {3, -1, -1}, {5, -1, -1}};

  // ---- carve workspace --------------------------------------------------
  char* wsp = (char*)d_ws;
  size_t off = 0;
  auto take = [&](size_t bytes) -> char* {
    char* p = wsp + off;
    off = (off + bytes + 255) & ~(size_t)255;
    return p;
  };
  size_t rowsTot = (size_t)pads[0] + pads[1] + pads[2] + pads[3];
  unsigned short* curBase = (unsigned short*)take(rowsTot * 256 * 2);
  float*          accBase = (float*)take(rowsTot * 256 * 4);
  unsigned short* curBF[4];
  float*          accF[4];
  {
    size_t o = 0;
    for (int t = 0; t < 4; ++t) {
      curBF[t] = curBase + o * 256;
      accF[t]  = accBase + o * 256;
      o += pads[t];
    }
  }
  float*          agg    = (float*)take((size_t)pads[0] * 256 * 4);
  float*          cnt    = (float*)take((size_t)pads[0] * 4);
  unsigned short* meanBF = (unsigned short*)take((size_t)pads[0] * 256 * 2);
  unsigned short* pWl    = (unsigned short*)take((size_t)14 * 65536 * 2);
  unsigned short* pWrc   = (unsigned short*)take((size_t)8 * 65536 * 2);
  unsigned short* pWout  = (unsigned short*)take((size_t)256 * 384 * 2);
  float*          bcomb  = (float*)take((size_t)8 * 256 * 4);

  // ---- pack weights (bf16 B-fragment layout) ----------------------------
  for (int l = 0; l < 2; ++l)
    for (int r = 0; r < 7; ++r)
      pack_w_kernel<<<256, 256, 0, stream>>>(
          Wl + (size_t)(l * 7 + r) * 65536, nullptr, nullptr, 256, 256,
          pWl + (size_t)(l * 7 + r) * 65536);

  for (int l = 0; l < 2; ++l)
    for (int t = 0; t < 4; ++t) {
      const float* w0 = Wr + (size_t)(l * 7 + relOfT[t][0]) * 65536;
      const float* w1 = relOfT[t][1] >= 0 ? Wr + (size_t)(l * 7 + relOfT[t][1]) * 65536 : nullptr;
      const float* w2 = relOfT[t][2] >= 0 ? Wr + (size_t)(l * 7 + relOfT[t][2]) * 65536 : nullptr;
      pack_w_kernel<<<256, 256, 0, stream>>>(w0, w1, w2, 256, 256,
                                             pWrc + (size_t)(l * 4 + t) * 65536);
      const float* b0 = bl + (size_t)(l * 7 + relOfT[t][0]) * 256;
      const float* b1 = relOfT[t][1] >= 0 ? bl + (size_t)(l * 7 + relOfT[t][1]) * 256 : nullptr;
      const float* b2 = relOfT[t][2] >= 0 ? bl + (size_t)(l * 7 + relOfT[t][2]) * 256 : nullptr;
      comb_bias_kernel<<<1, 256, 0, stream>>>(b0, b1, b2, bcomb + (l * 4 + t) * 256);
    }
  // W_out packed with Npad = 384 (multiple of 64) so GEMM loads need no guards
  pack_w_kernel<<<384, 256, 0, stream>>>(Wout, nullptr, nullptr, 349, 384, pWout);

  // ---- layer-0 inputs -> bf16 -------------------------------------------
  for (int t = 0; t < 4; ++t) {
    int n = nNodes[t] * 256;
    f32_to_bf16_kernel<<<(n + 255) / 256, 256, 0, stream>>>(xin[t], curBF[t], n);
  }

  // ---- 2 hetero SAGE layers ---------------------------------------------
  for (int l = 0; l < 2; ++l) {
    // self term: acc_t = x_t @ (sum Wr) + (sum bl)
    for (int t = 0; t < 4; ++t)
      gemm_bf16_kernel<<<dim3(pads[t] / 256, 4), 256, 0, stream>>>(
          curBF[t], pWrc + (size_t)(l * 4 + t) * 65536,
          bcomb + (l * 4 + t) * 256, accF[t], nNodes[t], 16, 256, 0);

    for (int r = 0; r < 7; ++r) {
      int st = relSrcT[r], dt = relDstT[r];
      int E = in_sizes[4 + 2 * r];
      const int* esrc = (const int*)d_in[4 + 2 * r];
      const int* edst = (const int*)d_in[5 + 2 * r];
      hipMemsetAsync(agg, 0, (size_t)nNodes[dt] * 256 * 4, stream);
      hipMemsetAsync(cnt, 0, (size_t)nNodes[dt] * 4, stream);
      gather_kernel<<<(E + 7) / 8, 256, 0, stream>>>(curBF[st], esrc, edst, E, agg, cnt);
      int n = nNodes[dt] * 256;
      mean_kernel<<<(n + 255) / 256, 256, 0, stream>>>(agg, cnt, meanBF, nNodes[dt]);
      gemm_bf16_kernel<<<dim3(pads[dt] / 256, 4), 256, 0, stream>>>(
          meanBF, pWl + (size_t)(l * 7 + r) * 65536, nullptr, accF[dt],
          nNodes[dt], 16, 256, 1);
    }

    for (int t = 0; t < 4; ++t) {
      int n = nNodes[t] * 256;
      relu_bf16_kernel<<<(n + 255) / 256, 256, 0, stream>>>(accF[t], curBF[t], n);
    }
  }

  // ---- output projection: p @ W_out + b_out  (N=349, Npad=384) ----------
  gemm_bf16_kernel<<<dim3(pads[0] / 256, 6), 256, 0, stream>>>(
      curBF[0], pWout, bout, (float*)d_out, nNodes[0], 24, 349, 0);
}
